// SparseConv_82566451298748
// MI455X (gfx1250) — matrix operand order repaired and checked by reading, not compiled, not run on hardware
//
#include <hip/hip_runtime.h>
#include <hip/hip_bf16.h>
#include <math.h>

// ---------------- CDNA5 WMMA types ----------------
typedef __attribute__((ext_vector_type(16))) _Float16 v16h;
typedef __attribute__((ext_vector_type(8)))  _Float16 v8h;
typedef __attribute__((ext_vector_type(8)))  float    v8f;

union AV { v16h v; v8h h[2]; };

// Problem constants (match reference)
#define CIN   64
#define COUT  64
#define K3    27
#define BENC  64
#define SHIFT 1
#define MAPN  (BENC*BENC*BENC)     // 262144 keys

// Swizzled weight buffer: [t][nt][ks][g][lane][8 halves]
//   q = (t*4+nt)*2+ks ; flat half index = ((q*2+g)*32 + lane)*8 + h2
#define KW_HALFS  (K3 * 4 * 2 * 2 * 32 * 8)   // 110,592 halves
#define KW_BYTES  (KW_HALFS * 2)              // 221,184 bytes (fits 320KB LDS)

// Workspace layout
#define WS_MAP_OFF    0
#define WS_MAP_BYTES  (MAPN * 4)
#define WS_FEAT_OFF   (WS_MAP_OFF + WS_MAP_BYTES)

#define WAVES_PER_BLOCK 16
#define TILES_PER_WAVE  2

// ---------------- kernel 1: init voxel map ----------------
__global__ void vmap_init_kernel(int* __restrict__ vmap) {
    int i = blockIdx.x * blockDim.x + threadIdx.x;
    if (i < MAPN) vmap[i] = -1;
}

// ---------------- kernel 2: scatter input indices ----------------
__global__ void vmap_scatter_kernel(const float* __restrict__ pos,
                                    const float* __restrict__ vsPtr,
                                    int* __restrict__ vmap, int n) {
    int i = blockIdx.x * blockDim.x + threadIdx.x;
    if (i >= n) return;
    float vs = vsPtr[0];
    int cx = (int)floorf(pos[i*3 + 0] / vs);
    int cy = (int)floorf(pos[i*3 + 1] / vs);
    int cz = (int)floorf(pos[i*3 + 2] / vs);
    int key = ((cz + SHIFT) * BENC + (cy + SHIFT)) * BENC + (cx + SHIFT);
    if (key >= 0 && key < MAPN) vmap[key] = i;
}

// ---------------- kernel 3: features f32 -> f16 (+ zero row at index N) ----
__global__ void feat_cvt_kernel(const float* __restrict__ f,
                                _Float16* __restrict__ fh, int n64, int total) {
    int i = blockIdx.x * blockDim.x + threadIdx.x;
    if (i < total) fh[i] = (i < n64) ? (_Float16)f[i] : (_Float16)0.0f;
}

// ---------------- kernel 4: weights -> B-register layout, LDS-friendly ----
// B 32x16 f16 register layout: col = lane&15; lanes 0-15 hold K 0..7,16..23,
// lanes 16-31 hold K 8..15,24..31; half h -> k_local = g*16 + (lane>>4)*8 + h2.
__global__ void kw_swizzle_kernel(const float* __restrict__ w,
                                  _Float16* __restrict__ kw) {
    int id = blockIdx.x * blockDim.x + threadIdx.x;   // K3*4*2*2*32 = 13824
    if (id >= K3 * 4 * 2 * 2 * 32) return;
    int lane = id & 31;
    int g    = (id >> 5) & 1;
    int ks   = (id >> 6) & 1;
    int nt   = (id >> 7) & 3;
    int t    =  id >> 9;
    int n    = nt * 16 + (lane & 15);
    int kb   = (lane >> 4) * 8;
    const float* src = w + (size_t)t * CIN * COUT;    // [cin][cout]
    _Float16* dst = kw + (size_t)id * 8;
    #pragma unroll
    for (int h2 = 0; h2 < 8; ++h2) {
        int k = ks * 32 + g * 16 + kb + h2;
        dst[h2] = (_Float16)src[k * COUT + n];
    }
}

// ---------------- kernel 5: main sparse-conv GEMM ----------------
// 512 threads = 16 waves; all weights staged in LDS; each wave does 2 M-tiles.
// Tap loop fully unrolled: straight-line code so the scheduler pipelines the
// random vmap / feature gathers across taps and renames A/B registers.
__global__ __launch_bounds__(512)
void sparse_conv_main_kernel(const float* __restrict__ outPos,
                             const float* __restrict__ vsPtr,
                             const _Float16* __restrict__ featsH,
                             const _Float16* __restrict__ kwH,
                             const int* __restrict__ vmap,
                             const float* __restrict__ bias,
                             float* __restrict__ out, int M, int zeroRow) {
    __shared__ __align__(16) _Float16 kwLDS[KW_HALFS];   // 221,184 B

    // ---- cooperative stage: global (linear, pre-swizzled) -> LDS ----
    {
        const float4* s = (const float4*)kwH;
        float4* d = (float4*)kwLDS;
        const int nvec = KW_BYTES / 16;                  // 13824
        for (int i = threadIdx.x; i < nvec; i += blockDim.x) d[i] = s[i];
    }
    __syncthreads();

    const int lane = threadIdx.x & 31;
    const int wave = threadIdx.x >> 5;
    const int numTiles = (M + 15) >> 4;
    const int tile0 = (blockIdx.x * WAVES_PER_BLOCK + wave) * TILES_PER_WAVE;
    if (tile0 >= numTiles) return;                       // wave-uniform

    const int col   = lane & 15;
    const int hi16  = lane >> 4;
    const int abase = hi16 * 8;                          // A K sub-base

    int row0 = tile0 * 16 + col;        if (row0 >= M) row0 = M - 1;
    int row1 = (tile0 + 1) * 16 + col;  if (row1 >= M) row1 = M - 1;

    const float vs = vsPtr[0];
    const int c0x = (int)floorf(outPos[row0*3 + 0] / vs);
    const int c0y = (int)floorf(outPos[row0*3 + 1] / vs);
    const int c0z = (int)floorf(outPos[row0*3 + 2] / vs);
    const int c1x = (int)floorf(outPos[row1*3 + 0] / vs);
    const int c1y = (int)floorf(outPos[row1*3 + 1] / vs);
    const int c1z = (int)floorf(outPos[row1*3 + 2] / vs);
    const int key0 = ((c0z + SHIFT) * BENC + (c0y + SHIFT)) * BENC + (c0x + SHIFT);
    const int key1 = ((c1z + SHIFT) * BENC + (c1y + SHIFT)) * BENC + (c1x + SHIFT);

    // 8 accumulators: [tile][nt]
    v8f acc[TILES_PER_WAVE][4];
    #pragma unroll
    for (int nt = 0; nt < 4; ++nt) {
        float b = bias[nt * 16 + col];
        #pragma unroll
        for (int i = 0; i < 8; ++i) { acc[0][nt][i] = b; acc[1][nt][i] = b; }
    }

    #pragma unroll
    for (int t = 0; t < K3; ++t) {
        // constant tap offset (folds at compile time under full unroll)
        const int dd = ((t / 9 - 1) * BENC + ((t / 3) % 3 - 1)) * BENC + (t % 3 - 1);

        int i0 = vmap[key0 + dd];  i0 = (i0 < 0) ? zeroRow : i0;   // miss -> zero row
        int i1 = vmap[key1 + dd];  i1 = (i1 < 0) ? zeroRow : i1;
        const _Float16* fp0 = featsH + (size_t)i0 * CIN;
        const _Float16* fp1 = featsH + (size_t)i1 * CIN;

        // A tiles: [tile][ks], unconditional aligned 16B loads
        AV a0k0, a0k1, a1k0, a1k1;
        a0k0.h[0] = *(const v8h*)(fp0 +  0 + abase);
        a0k0.h[1] = *(const v8h*)(fp0 + 16 + abase);
        a0k1.h[0] = *(const v8h*)(fp0 + 32 + abase);
        a0k1.h[1] = *(const v8h*)(fp0 + 48 + abase);
        a1k0.h[0] = *(const v8h*)(fp1 +  0 + abase);
        a1k0.h[1] = *(const v8h*)(fp1 + 16 + abase);
        a1k1.h[0] = *(const v8h*)(fp1 + 32 + abase);
        a1k1.h[1] = *(const v8h*)(fp1 + 48 + abase);

        #pragma unroll
        for (int ks = 0; ks < 2; ++ks) {
            const v16h av0 = (ks == 0) ? a0k0.v : a0k1.v;
            const v16h av1 = (ks == 0) ? a1k0.v : a1k1.v;
            #pragma unroll
            for (int nt = 0; nt < 4; ++nt) {
                const int q = (t * 4 + nt) * 2 + ks;
                AV b;
                b.h[0] = *(const v8h*)(kwLDS + ((size_t)(q * 2 + 0) * 32 + lane) * 8);
                b.h[1] = *(const v8h*)(kwLDS + ((size_t)(q * 2 + 1) * 32 + lane) * 8);
                acc[0][nt] = __builtin_amdgcn_wmma_f32_16x16x32_f16(
                    false, av0, false, b.v, (short)0, acc[0][nt], false, false);
                acc[1][nt] = __builtin_amdgcn_wmma_f32_16x16x32_f16(
                    false, av1, false, b.v, (short)0, acc[1][nt], false, false);
            }
        }
    }

    // D layout: lane covers col = lane&15, rows m = i + 8*(lane>>4)
    #pragma unroll
    for (int tt = 0; tt < TILES_PER_WAVE; ++tt) {
        const int tile = tile0 + tt;
        float* obase = out + (size_t)tile * 16 * COUT;
        #pragma unroll
        for (int i = 0; i < 8; ++i) {
            int m = i + hi16 * 8;
            if (tile * 16 + m < M) {
                float* orow = obase + (size_t)m * COUT + col;
                orow[ 0] = acc[tt][0][i];
                orow[16] = acc[tt][1][i];
                orow[32] = acc[tt][2][i];
                orow[48] = acc[tt][3][i];
            }
        }
    }
}

// ---------------- host-side launcher ----------------
extern "C" void kernel_launch(void* const* d_in, const int* in_sizes, int n_in,
                              void* d_out, int out_size, void* d_ws, size_t ws_size,
                              hipStream_t stream) {
    const float* inp_features  = (const float*)d_in[0];
    const float* inp_positions = (const float*)d_in[1];
    const float* out_positions = (const float*)d_in[2];
    const float* voxel_size    = (const float*)d_in[3];
    const float* kernel        = (const float*)d_in[4];
    const float* bias          = (const float*)d_in[5];

    const int N = in_sizes[0] / CIN;
    const int M = in_sizes[2] / 3;

    char* ws = (char*)d_ws;
    int*      vmap   = (int*)(ws + WS_MAP_OFF);
    _Float16* featsH = (_Float16*)(ws + WS_FEAT_OFF);
    size_t feat_bytes = (size_t)(N + 1) * CIN * 2;        // +1 zero row
    size_t kw_off = (WS_FEAT_OFF + feat_bytes + 255) & ~(size_t)255;
    _Float16* kwH = (_Float16*)(ws + kw_off);

    float* out = (float*)d_out;

    vmap_init_kernel<<<(MAPN + 255) / 256, 256, 0, stream>>>(vmap);
    vmap_scatter_kernel<<<(N + 255) / 256, 256, 0, stream>>>(
        inp_positions, voxel_size, vmap, N);
    const int n64 = N * CIN;
    const int tot64 = (N + 1) * CIN;
    feat_cvt_kernel<<<(tot64 + 255) / 256, 256, 0, stream>>>(
        inp_features, featsH, n64, tot64);
    kw_swizzle_kernel<<<(K3 * 4 * 2 * 2 * 32 + 255) / 256, 256, 0, stream>>>(
        kernel, kwH);

    const int numTiles = (M + 15) / 16;
    const int tilesPerBlock = WAVES_PER_BLOCK * TILES_PER_WAVE;   // 32
    const int grid = (numTiles + tilesPerBlock - 1) / tilesPerBlock;
    sparse_conv_main_kernel<<<grid, WAVES_PER_BLOCK * 32, 0, stream>>>(
        out_positions, voxel_size, featsH, kwH, vmap, bias, out, M, N);
}
